// DNA_18468359373122
// MI455X (gfx1250) — matrix operand order repaired and checked by reading, not compiled, not run on hardware
//
#include <hip/hip_runtime.h>

// ---------------------------------------------------------------------------
// MI455X (gfx1250) implementation: bf16 WMMA GEMMs + flash attention.
// wave32 everywhere. All matrix math via v_wmma_f32_16x16x32_bf16.
// ---------------------------------------------------------------------------

typedef __attribute__((ext_vector_type(16))) __bf16        v16bf;
typedef __attribute__((ext_vector_type(8)))  float         v8f;
typedef __attribute__((ext_vector_type(4)))  float         v4f;
typedef __attribute__((ext_vector_type(4)))  unsigned int  v4u;

union Frag16 { v16bf v; v4u q[2]; };

static constexpr int T_   = 2048;
static constexpr int D_   = 512;
static constexpr int H_   = 8;
static constexpr int DH_  = 64;
static constexpr int CAP_ = 512;
static constexpr int NE_  = 8;      // trained experts (expert 8 is the null expert)
static constexpr int QKV_ = 1536;
static constexpr int FF_  = 2048;

static __device__ __forceinline__ unsigned short f2bf(float f) {
  unsigned int u = __float_as_uint(f);
  unsigned int r = u + 0x7FFFu + ((u >> 16) & 1u);   // round-to-nearest-even
  return (unsigned short)(r >> 16);
}

// ---------------------------------------------------------------- utilities
__global__ void k_f32_to_bf16(const float* __restrict__ in, unsigned short* __restrict__ out, int n) {
  int i = blockIdx.x * 256 + threadIdx.x;
  if (i < n) out[i] = f2bf(in[i]);
}

__global__ void k_embed(const int* __restrict__ ids, const float* __restrict__ W, float* __restrict__ h) {
  int t = blockIdx.x, tid = threadIdx.x;                 // block 128
  int row = ids[t];
  ((v4f*)h)[(size_t)t * 128 + tid] = ((const v4f*)W)[(size_t)row * 128 + tid];
}

__global__ void k_rope(float* __restrict__ cosb, float* __restrict__ sinb) {
  int t = blockIdx.x, j = threadIdx.x;                   // block 64
  int i = j & 31;
  float inv = __powf(10000.0f, -(float)i / 32.0f);
  float f = (float)t * inv;
  cosb[t * 64 + j] = cosf(f);
  sinb[t * 64 + j] = sinf(f);
}

__global__ void k_rmsnorm_bf16(const float* __restrict__ x, const float* __restrict__ s,
                               unsigned short* __restrict__ out) {
  int row = blockIdx.x, tid = threadIdx.x;               // block 256, D=512
  const float* xr = x + (size_t)row * D_;
  float a = xr[tid], b = xr[tid + 256];
  __shared__ float red[256];
  red[tid] = a * a + b * b;
  __syncthreads();
  for (int o = 128; o > 0; o >>= 1) { if (tid < o) red[tid] += red[tid + o]; __syncthreads(); }
  float inv = rsqrtf(red[0] / (float)D_ + 1e-6f);
  out[(size_t)row * D_ + tid]       = f2bf(a * inv * s[tid]);
  out[(size_t)row * D_ + tid + 256] = f2bf(b * inv * s[tid + 256]);
}

__global__ void k_rmsnorm_f32(const float* __restrict__ x, const float* __restrict__ s,
                              float* __restrict__ out) {
  int row = blockIdx.x, tid = threadIdx.x;
  const float* xr = x + (size_t)row * D_;
  float a = xr[tid], b = xr[tid + 256];
  __shared__ float red[256];
  red[tid] = a * a + b * b;
  __syncthreads();
  for (int o = 128; o > 0; o >>= 1) { if (tid < o) red[tid] += red[tid + o]; __syncthreads(); }
  float inv = rsqrtf(red[0] / (float)D_ + 1e-6f);
  out[(size_t)row * D_ + tid]       = a * inv * s[tid];
  out[(size_t)row * D_ + tid + 256] = b * inv * s[tid + 256];
}

__global__ void k_gelu_bf16(const float* __restrict__ in, unsigned short* __restrict__ out, int n) {
  int i = blockIdx.x * 256 + threadIdx.x;
  if (i < n) {
    float x = in[i];
    float t = tanhf(0.7978845608028654f * (x + 0.044715f * x * x * x));
    out[i] = f2bf(0.5f * x * (1.0f + t));
  }
}

// ------------------------------------------------------------- WMMA NT-GEMM
// C[M,N](f32) (+)= A[M,K](bf16 row-major) * B[N,K](bf16 row-major)^T
// block = 128 (4 waves); wave computes a 16x64 tile; grid = (N/64, M/64).
__global__ void k_gemm_nt(const unsigned short* __restrict__ A,
                          const unsigned short* __restrict__ B,
                          float* __restrict__ C,
                          int M, int N, int K, int accumulate) {
  const int lane = threadIdx.x & 31;
  const int wave = threadIdx.x >> 5;
  const int lm = lane & 15, lh = lane >> 4;
  const int m0 = blockIdx.y * 64 + wave * 16;
  const int n0 = blockIdx.x * 64;
  if (m0 >= M) return;

  v8f acc[4];
  for (int i = 0; i < 4; i++) for (int r = 0; r < 8; r++) acc[i][r] = 0.0f;

  const unsigned short* arow = A + (size_t)(m0 + lm) * K;
  for (int k0 = 0; k0 < K; k0 += 32) {
    Frag16 fa;
    fa.q[0] = *(const v4u*)(arow + k0 + lh * 16);
    fa.q[1] = *(const v4u*)(arow + k0 + lh * 16 + 8);
    __builtin_prefetch(arow + k0 + 128, 0, 1);
#pragma unroll
    for (int nt = 0; nt < 4; nt++) {
      const unsigned short* brow = B + (size_t)(n0 + nt * 16 + lm) * K + k0 + lh * 16;
      Frag16 fb;
      fb.q[0] = ((const v4u*)brow)[0];
      fb.q[1] = ((const v4u*)brow)[1];
      acc[nt] = __builtin_amdgcn_wmma_f32_16x16x32_bf16(false, fa.v, false, fb.v,
                                                        (short)0, acc[nt], false, false);
    }
  }
  for (int nt = 0; nt < 4; nt++) {
    const int n = n0 + nt * 16 + lm;
    for (int r = 0; r < 8; r++) {
      const int m = m0 + r + lh * 8;
      size_t off = (size_t)m * N + n;
      float vv = acc[nt][r];
      C[off] = accumulate ? (C[off] + vv) : vv;
    }
  }
}

// ----------------------------------------------- QKV post: RoPE + layouting
// qkv f32 [ctx][1536] -> Q,K bf16 [H][ctx][64] (Q pre-scaled 1/8), Vt bf16 [H][64][ctx]
__global__ void k_qkv_post(const float* __restrict__ qkv,
                           const float* __restrict__ cosb, const float* __restrict__ sinb,
                           unsigned short* __restrict__ Q, unsigned short* __restrict__ K,
                           unsigned short* __restrict__ Vt, int ctx) {
  int t = blockIdx.x;
  int hd = threadIdx.x >> 5, j = threadIdx.x & 31;       // block 256 = 8 heads * 32
  const float* row = qkv + (size_t)t * QKV_;
  float c1 = cosb[t * 64 + j],      s1 = sinb[t * 64 + j];
  float c2 = cosb[t * 64 + 32 + j], s2 = sinb[t * 64 + 32 + j];
  float q1 = row[hd * 64 + j],        q2 = row[hd * 64 + 32 + j];
  float k1 = row[512 + hd * 64 + j],  k2 = row[512 + hd * 64 + 32 + j];
  float v1 = row[1024 + hd * 64 + j], v2 = row[1024 + hd * 64 + 32 + j];
  float qr1 = q1 * c1 - q2 * s1, qr2 = q2 * c2 + q1 * s2;
  float kr1 = k1 * c1 - k2 * s1, kr2 = k2 * c2 + k1 * s2;
  const float sc = 0.125f;                               // 1/sqrt(dh)
  size_t base = ((size_t)hd * ctx + t) * 64;
  Q[base + j]      = f2bf(qr1 * sc);
  Q[base + 32 + j] = f2bf(qr2 * sc);
  K[base + j]      = f2bf(kr1);
  K[base + 32 + j] = f2bf(kr2);
  Vt[((size_t)hd * 64 + j) * ctx + t]      = f2bf(v1);
  Vt[((size_t)hd * 64 + 32 + j) * ctx + t] = f2bf(v2);
}

// ------------------------------------------------------ flash attention (WMMA)
// one wave per 16-query tile; online softmax; P staged through LDS.
__global__ void k_attn(const unsigned short* __restrict__ Q,
                       const unsigned short* __restrict__ K,
                       const unsigned short* __restrict__ Vt,
                       unsigned short* __restrict__ O, int ctx) {
  __shared__ __align__(16) unsigned short pst[4][16 * 32];
  const int lane = threadIdx.x & 31;
  const int wave = threadIdx.x >> 5;
  const int lm = lane & 15, lh = lane >> 4;
  const int hd = blockIdx.y;
  const int q0 = blockIdx.x * 64 + wave * 16;
  const unsigned short* Qh = Q + (size_t)hd * ctx * DH_;
  const unsigned short* Kh = K + (size_t)hd * ctx * DH_;
  const unsigned short* Vh = Vt + (size_t)hd * DH_ * ctx;
  unsigned short* P = pst[wave];

  Frag16 fq[2];
  const unsigned short* qrow = Qh + (size_t)(q0 + lm) * DH_;
#pragma unroll
  for (int d = 0; d < 2; d++) {
    fq[d].q[0] = *(const v4u*)(qrow + d * 32 + lh * 16);
    fq[d].q[1] = *(const v4u*)(qrow + d * 32 + lh * 16 + 8);
  }
  v8f o[4];
  for (int i = 0; i < 4; i++) for (int r = 0; r < 8; r++) o[i][r] = 0.0f;
  float mrow[8], lrow[8];
  for (int r = 0; r < 8; r++) { mrow[r] = -__builtin_inff(); lrow[r] = 0.0f; }

  for (int kb = 0; kb < q0 + 16; kb += 32) {             // causal: only needed k-tiles
    v8f s[2];
    for (int nt = 0; nt < 2; nt++) for (int r = 0; r < 8; r++) s[nt][r] = 0.0f;
#pragma unroll
    for (int nt = 0; nt < 2; nt++) {
#pragma unroll
      for (int d = 0; d < 2; d++) {
        const unsigned short* krow = Kh + (size_t)(kb + nt * 16 + lm) * DH_ + d * 32 + lh * 16;
        Frag16 fb; fb.q[0] = ((const v4u*)krow)[0]; fb.q[1] = ((const v4u*)krow)[1];
        s[nt] = __builtin_amdgcn_wmma_f32_16x16x32_bf16(false, fq[d].v, false, fb.v,
                                                        (short)0, s[nt], false, false);
      }
    }
#pragma unroll
    for (int r = 0; r < 8; r++) {
      const int qq = q0 + r + 8 * lh;
      if (kb + lm > qq)      s[0][r] = -1e9f;            // causal mask (matches ref -1e9)
      if (kb + 16 + lm > qq) s[1][r] = -1e9f;
      float v = fmaxf(s[0][r], s[1][r]);
#pragma unroll
      for (int off = 1; off < 16; off <<= 1) v = fmaxf(v, __shfl_xor(v, off, 32));
      float mnew = fmaxf(mrow[r], v);
      float scale = __expf(mrow[r] - mnew);
      mrow[r] = mnew;
      float p0 = __expf(s[0][r] - mnew);
      float p1 = __expf(s[1][r] - mnew);
      P[(r + 8 * lh) * 32 + lm]      = f2bf(p0);
      P[(r + 8 * lh) * 32 + 16 + lm] = f2bf(p1);
      float ps = p0 + p1;
#pragma unroll
      for (int off = 1; off < 16; off <<= 1) ps += __shfl_xor(ps, off, 32);
      lrow[r] = lrow[r] * scale + ps;
      for (int i = 0; i < 4; i++) o[i][r] *= scale;
    }
    Frag16 fp;                                           // re-read P in A-fragment layout
    fp.q[0] = *(const v4u*)(P + lm * 32 + lh * 16);
    fp.q[1] = *(const v4u*)(P + lm * 32 + lh * 16 + 8);
#pragma unroll
    for (int i = 0; i < 4; i++) {
      const unsigned short* vrow = Vh + (size_t)(i * 16 + lm) * ctx + kb + lh * 16;
      Frag16 fb; fb.q[0] = ((const v4u*)vrow)[0]; fb.q[1] = ((const v4u*)vrow)[1];
      o[i] = __builtin_amdgcn_wmma_f32_16x16x32_bf16(false, fp.v, false, fb.v,
                                                     (short)0, o[i], false, false);
    }
  }
  for (int i = 0; i < 4; i++)
    for (int r = 0; r < 8; r++) {
      float val = o[i][r] / lrow[r];
      O[(size_t)(q0 + r + 8 * lh) * (H_ * DH_) + hd * DH_ + i * 16 + lm] = f2bf(val);
    }
}

// ------------------------------------------------------------------ routing
__global__ void k_router(const float* __restrict__ h, const float* __restrict__ wr,
                         float* __restrict__ g, float* __restrict__ cw,
                         int* __restrict__ slot_of, int* __restrict__ top2e) {
  int t = blockIdx.x, lane = threadIdx.x;                // block 32 (one wave)
  float lg[9];
  const float* hr = h + (size_t)t * D_;
  for (int e = 0; e < 9; e++) {
    const float* w = wr + e * D_;
    float s = 0.0f;
    for (int d = lane; d < D_; d += 32) s += hr[d] * w[d];
    for (int o = 16; o > 0; o >>= 1) s += __shfl_xor(s, o, 32);
    lg[e] = s;
  }
  float mx = lg[0];
  for (int e = 1; e < 9; e++) mx = fmaxf(mx, lg[e]);
  float p[9], den = 0.0f;
  for (int e = 0; e < 9; e++) { p[e] = __expf(lg[e] - mx); den += p[e]; }
  for (int e = 0; e < 9; e++) p[e] /= den;
  int i1 = 0;
  for (int e = 1; e < 9; e++) if (lg[e] > lg[i1]) i1 = e;      // first max (ties -> low idx)
  int i2 = -1;
  for (int e = 0; e < 9; e++) { if (e == i1) continue; if (i2 < 0 || lg[e] > lg[i2]) i2 = e; }
  if (lane < NE_) {
    int e = lane;
    g[(size_t)e * T_ + t] = (e == i1 || e == i2) ? p[e] : -__builtin_inff();
    cw[(size_t)e * T_ + t] = 0.0f;
    slot_of[(size_t)e * T_ + t] = -1;
  }
  if (lane == 0) { top2e[2 * t] = i1; top2e[2 * t + 1] = i2; }
}

// top-CAP per expert: in-LDS bitonic sort of packed (prob,idx) keys, desc.
__global__ void k_select(const float* __restrict__ g, float* __restrict__ cw,
                         int* __restrict__ slot_of, int* __restrict__ slot_token) {
  __shared__ unsigned long long keys[T_];
  int e = blockIdx.x, tid = threadIdx.x;                 // block 512
  for (int i = tid; i < T_; i += 512) {
    float v = g[(size_t)e * T_ + i];
    unsigned int u = __float_as_uint(v);
    unsigned int ou = (u & 0x80000000u) ? ~u : (u | 0x80000000u);
    keys[i] = ((unsigned long long)ou << 32) | (unsigned int)(~i);
  }
  __syncthreads();
  for (unsigned k = 2; k <= (unsigned)T_; k <<= 1) {
    for (unsigned j = k >> 1; j > 0; j >>= 1) {
      for (unsigned i = tid; i < (unsigned)T_; i += 512) {
        unsigned ixj = i ^ j;
        if (ixj > i) {
          unsigned long long a = keys[i], b = keys[ixj];
          bool up = ((i & k) == 0);
          if (up ? (a < b) : (a > b)) { keys[i] = b; keys[ixj] = a; }
        }
      }
      __syncthreads();
    }
  }
  if (tid < CAP_) {
    unsigned long long kk = keys[tid];
    unsigned int ou = (unsigned int)(kk >> 32);
    int tok = (int)(~(unsigned int)kk);
    if (ou & 0x80000000u) {                              // finite (prob >= 0) -> valid slot
      slot_token[e * CAP_ + tid] = tok;
      slot_of[(size_t)e * T_ + tok] = tid;
      cw[(size_t)e * T_ + tok] = __uint_as_float(ou ^ 0x80000000u);
    } else {
      slot_token[e * CAP_ + tid] = -1;
    }
  }
}

__global__ void k_gather(const int* __restrict__ slot_token, const float* __restrict__ h,
                         const float* __restrict__ cosb, const float* __restrict__ sinb,
                         float* __restrict__ xin, float* __restrict__ eo,
                         float* __restrict__ cos_r, float* __restrict__ sin_r) {
  int e = blockIdx.x, c = blockIdx.y, tid = threadIdx.x; // block 128
  int tok = slot_token[e * CAP_ + c];
  size_t sl = (size_t)e * CAP_ + c;
  v4f z; z.x = 0.f; z.y = 0.f; z.z = 0.f; z.w = 0.f;
  v4f v = (tok >= 0) ? ((const v4f*)h)[(size_t)tok * 128 + tid] : z;
  ((v4f*)xin)[sl * 128 + tid] = v;
  ((v4f*)eo)[sl * 128 + tid] = v;                        // residual base for experts
  if (tid < 64) {
    cos_r[sl * 64 + tid] = (tok >= 0) ? cosb[tok * 64 + tid] : 0.0f;
    sin_r[sl * 64 + tid] = (tok >= 0) ? sinb[tok * 64 + tid] : 0.0f;
  }
}

__global__ void k_combine(float* __restrict__ h, const float* __restrict__ eo,
                          const float* __restrict__ cw, const int* __restrict__ slot_of,
                          const int* __restrict__ top2e) {
  int t = blockIdx.x, tid = threadIdx.x;                 // block 128
  int e1 = top2e[2 * t], e2 = top2e[2 * t + 1];
  float w1 = 0.0f, w2 = 0.0f;
  int s1 = -1, s2 = -1;
  if (e1 < NE_) { s1 = slot_of[(size_t)e1 * T_ + t]; if (s1 >= 0) w1 = cw[(size_t)e1 * T_ + t]; }
  if (e2 < NE_) { s2 = slot_of[(size_t)e2 * T_ + t]; if (s2 >= 0) w2 = cw[(size_t)e2 * T_ + t]; }
  float rho = w1 + w2;
  v4f hv = ((const v4f*)h)[(size_t)t * 128 + tid];
  v4f acc = hv * (1.0f - rho);
  if (s1 >= 0) acc += ((const v4f*)eo)[((size_t)e1 * CAP_ + s1) * 128 + tid] * w1;
  if (s2 >= 0) acc += ((const v4f*)eo)[((size_t)e2 * CAP_ + s2) * 128 + tid] * w2;
  ((v4f*)h)[(size_t)t * 128 + tid] = acc;
}

// ---------------------------------------------------------------------------
extern "C" void kernel_launch(void* const* d_in, const int* in_sizes, int n_in,
                              void* d_out, int out_size, void* d_ws, size_t ws_size,
                              hipStream_t stream) {
  (void)in_sizes; (void)n_in; (void)out_size;
  const int*   ids      = (const int*)  d_in[0];
  const float* embed_w  = (const float*)d_in[1];
  const float* router_w = (const float*)d_in[2];
  const float* attn_qkv = (const float*)d_in[3];
  const float* attn_o   = (const float*)d_in[4];
  const float* attn_ln  = (const float*)d_in[5];
  const float* ffn_w1   = (const float*)d_in[6];
  const float* ffn_w2   = (const float*)d_in[7];
  const float* ffn_ln   = (const float*)d_in[8];
  const float* bb_qkv   = (const float*)d_in[9];
  const float* bb_o     = (const float*)d_in[10];
  const float* bb_ln    = (const float*)d_in[11];
  const float* lnout_s  = (const float*)d_in[12];
  float* out = (float*)d_out;

  char* p0 = (char*)d_ws;
  char* p = p0;
  auto alloc = [&](size_t bytes) -> char* {
    char* r = p; p += (bytes + 255) & ~(size_t)255; return r;
  };

  float* h     = (float*)alloc((size_t)T_ * D_ * 4);
  float* cosb  = (float*)alloc((size_t)T_ * DH_ * 4);
  float* sinb  = (float*)alloc((size_t)T_ * DH_ * 4);
  unsigned short* bbqkv_bf = (unsigned short*)alloc((size_t)QKV_ * D_ * 2);
  unsigned short* bbo_bf   = (unsigned short*)alloc((size_t)D_ * D_ * 2);
  unsigned short* aqkv_bf  = (unsigned short*)alloc((size_t)4 * QKV_ * D_ * 2);
  unsigned short* ao_bf    = (unsigned short*)alloc((size_t)4 * D_ * D_ * 2);
  unsigned short* w1_bf    = (unsigned short*)alloc((size_t)4 * FF_ * D_ * 2);
  unsigned short* w2_bf    = (unsigned short*)alloc((size_t)4 * D_ * FF_ * 2);
  unsigned short* u_bb = (unsigned short*)alloc((size_t)T_ * D_ * 2);
  unsigned short* u_ex = (unsigned short*)alloc((size_t)NE_ * CAP_ * D_ * 2);
  float* big = (float*)alloc((size_t)4 * CAP_ * FF_ * 4);     // qkv / ffn-hidden scratch
  unsigned short* Qb  = (unsigned short*)alloc((size_t)H_ * T_ * DH_ * 2);
  unsigned short* Kb  = (unsigned short*)alloc((size_t)H_ * T_ * DH_ * 2);
  unsigned short* Vtb = (unsigned short*)alloc((size_t)H_ * DH_ * T_ * 2);
  unsigned short* Qe  = (unsigned short*)alloc((size_t)4 * H_ * CAP_ * DH_ * 2);
  unsigned short* Ke  = (unsigned short*)alloc((size_t)4 * H_ * CAP_ * DH_ * 2);
  unsigned short* Vte = (unsigned short*)alloc((size_t)4 * H_ * DH_ * CAP_ * 2);
  unsigned short* Obb = (unsigned short*)alloc((size_t)T_ * D_ * 2);
  unsigned short* Oe  = (unsigned short*)alloc((size_t)4 * CAP_ * D_ * 2);
  unsigned short* gbf = (unsigned short*)alloc((size_t)4 * CAP_ * FF_ * 2);
  float* xin  = (float*)alloc((size_t)NE_ * CAP_ * D_ * 4);
  float* eo   = (float*)alloc((size_t)NE_ * CAP_ * D_ * 4);
  float* cosr = (float*)alloc((size_t)NE_ * CAP_ * DH_ * 4);
  float* sinr = (float*)alloc((size_t)NE_ * CAP_ * DH_ * 4);
  float* g    = (float*)alloc((size_t)NE_ * T_ * 4);
  float* cw   = (float*)alloc((size_t)NE_ * T_ * 4);
  int*   slot_of    = (int*)alloc((size_t)NE_ * T_ * 4);
  int*   top2e      = (int*)alloc((size_t)T_ * 2 * 4);
  int*   slot_token = (int*)alloc((size_t)NE_ * CAP_ * 4);
  if ((size_t)(p - p0) > ws_size) return;                 // insufficient scratch: bail safely

  auto cvt = [&](const float* src, unsigned short* dst, size_t n) {
    k_f32_to_bf16<<<dim3((unsigned)((n + 255) / 256)), 256, 0, stream>>>(src, dst, (int)n);
  };
  cvt(bb_qkv,   bbqkv_bf, (size_t)QKV_ * D_);
  cvt(bb_o,     bbo_bf,   (size_t)D_ * D_);
  cvt(attn_qkv, aqkv_bf,  (size_t)4 * QKV_ * D_);
  cvt(attn_o,   ao_bf,    (size_t)4 * D_ * D_);
  cvt(ffn_w1,   w1_bf,    (size_t)4 * FF_ * D_);
  cvt(ffn_w2,   w2_bf,    (size_t)4 * D_ * FF_);

  k_embed<<<T_, 128, 0, stream>>>(ids, embed_w, h);
  k_rope<<<T_, 64, 0, stream>>>(cosb, sinb);

  // ---------------- backbone attention block
  k_rmsnorm_bf16<<<T_, 256, 0, stream>>>(h, bb_ln, u_bb);
  k_gemm_nt<<<dim3(QKV_ / 64, T_ / 64), 128, 0, stream>>>(u_bb, bbqkv_bf, big, T_, QKV_, D_, 0);
  k_qkv_post<<<T_, 256, 0, stream>>>(big, cosb, sinb, Qb, Kb, Vtb, T_);
  k_attn<<<dim3(T_ / 64, H_), 128, 0, stream>>>(Qb, Kb, Vtb, Obb, T_);
  k_gemm_nt<<<dim3(D_ / 64, T_ / 64), 128, 0, stream>>>(Obb, bbo_bf, h, T_, D_, D_, 1);

  // ---------------- MoE hops
  for (int hop = 0; hop < 2; hop++) {
    k_router<<<T_, 32, 0, stream>>>(h, router_w + (size_t)hop * 9 * D_, g, cw, slot_of, top2e);
    k_select<<<NE_, 512, 0, stream>>>(g, cw, slot_of, slot_token);
    k_gather<<<dim3(NE_, CAP_), 128, 0, stream>>>(slot_token, h, cosb, sinb, xin, eo, cosr, sinr);

    for (int a = 0; a < 4; a++) {                         // attention experts e = 0,2,4,6
      int e = 2 * a;
      const float* xe = xin + (size_t)e * CAP_ * D_;
      unsigned short* ue = u_ex + (size_t)e * CAP_ * D_;
      float* qkvb = big + (size_t)a * CAP_ * QKV_;
      unsigned short* Qa = Qe + (size_t)a * H_ * CAP_ * DH_;
      unsigned short* Ka = Ke + (size_t)a * H_ * CAP_ * DH_;
      unsigned short* Va = Vte + (size_t)a * H_ * DH_ * CAP_;
      unsigned short* Oa = Oe + (size_t)a * CAP_ * D_;
      k_rmsnorm_bf16<<<CAP_, 256, 0, stream>>>(xe, attn_ln + a * D_, ue);
      k_gemm_nt<<<dim3(QKV_ / 64, CAP_ / 64), 128, 0, stream>>>(ue, aqkv_bf + (size_t)a * QKV_ * D_,
                                                                qkvb, CAP_, QKV_, D_, 0);
      k_qkv_post<<<CAP_, 256, 0, stream>>>(qkvb, cosr + (size_t)e * CAP_ * DH_,
                                           sinr + (size_t)e * CAP_ * DH_, Qa, Ka, Va, CAP_);
      k_attn<<<dim3(CAP_ / 64, H_), 128, 0, stream>>>(Qa, Ka, Va, Oa, CAP_);
      k_gemm_nt<<<dim3(D_ / 64, CAP_ / 64), 128, 0, stream>>>(Oa, ao_bf + (size_t)a * D_ * D_,
                                                              eo + (size_t)e * CAP_ * D_, CAP_, D_, D_, 1);
    }
    for (int f = 0; f < 4; f++) {                         // FFN experts e = 1,3,5,7
      int e = 2 * f + 1;
      const float* xe = xin + (size_t)e * CAP_ * D_;
      unsigned short* ue = u_ex + (size_t)e * CAP_ * D_;
      float* gfb = big + (size_t)f * CAP_ * FF_;
      unsigned short* gb = gbf + (size_t)f * CAP_ * FF_;
      k_rmsnorm_bf16<<<CAP_, 256, 0, stream>>>(xe, ffn_ln + f * D_, ue);
      k_gemm_nt<<<dim3(FF_ / 64, CAP_ / 64), 128, 0, stream>>>(ue, w1_bf + (size_t)f * FF_ * D_,
                                                               gfb, CAP_, FF_, D_, 0);
      k_gelu_bf16<<<(CAP_ * FF_) / 256, 256, 0, stream>>>(gfb, gb, CAP_ * FF_);
      k_gemm_nt<<<dim3(D_ / 64, CAP_ / 64), 128, 0, stream>>>(gb, w2_bf + (size_t)f * D_ * FF_,
                                                              eo + (size_t)e * CAP_ * D_, CAP_, D_, FF_, 1);
    }
    k_combine<<<T_, 128, 0, stream>>>(h, eo, cw, slot_of, top2e);
  }

  k_rmsnorm_f32<<<T_, 256, 0, stream>>>(h, lnout_s, out);
}
